// FlashscGPTLayer_27779848470549
// MI455X (gfx1250) — matrix operand order repaired and checked by reading, not compiled, not run on hardware
//
#include <hip/hip_runtime.h>
#include <cstdint>

// ---------------------------------------------------------------------------
// CDNA5 (gfx1250) bf16-WMMA implementation of one scGPT transformer layer.
// wave32; v_wmma_f32_16x16x32_bf16 for all matmuls; GLOBAL_LOAD_ASYNC_TO_LDS
// (ASYNCcnt) + double-buffered LDS tiles for data movement.
// ---------------------------------------------------------------------------

typedef __attribute__((ext_vector_type(16))) __bf16 bf16x16;
typedef __attribute__((ext_vector_type(8)))  float  f32x8;
typedef __attribute__((ext_vector_type(4)))  int    i32x4;

#define WMMA_BF16(a, b, c) \
    __builtin_amdgcn_wmma_f32_16x16x32_bf16(false, (a), false, (b), (short)0, (c), false, false)

#define F32X8_ZERO {0.f,0.f,0.f,0.f,0.f,0.f,0.f,0.f}

#if __has_builtin(__builtin_amdgcn_global_load_async_to_lds_b128) && \
    __has_builtin(__builtin_amdgcn_s_wait_asynccnt)
#define HAVE_ASYNC_LDS 1
#else
#define HAVE_ASYNC_LDS 0
#endif

// Copy 16 bytes global -> LDS. Async path: GLOBAL_LOAD_ASYNC_TO_LDS_B128
// (no VGPR round-trip, tracked by ASYNCcnt). Fallback: register round-trip.
// Builtin signature (from compiler diagnostic): params are int4* pointers.
__device__ __forceinline__ void copy16_g2l(void* lds_dst, const void* gsrc) {
#if HAVE_ASYNC_LDS
    __builtin_amdgcn_global_load_async_to_lds_b128(
        (i32x4*)gsrc, (i32x4*)lds_dst, /*offset=*/0, /*cpol=*/0);
#else
    *(uint4*)lds_dst = *(const uint4*)gsrc;
#endif
}

__device__ __forceinline__ void async_wait0() {
#if HAVE_ASYNC_LDS
    __builtin_amdgcn_s_wait_asynccnt(0);
#endif
}

// A-matrix (16x32, 16-bit) per-lane K index (contiguous order, matching B fills):
//   lanes 0-15 K=e ; lanes 16-31 K=16+e
__device__ __forceinline__ int a_frag_k(int half, int e) {
    return half * 16 + e;
}

// ---------------------------------------------------------------------------
// f32 -> bf16 conversion (weights / activation staging)
// ---------------------------------------------------------------------------
__global__ __launch_bounds__(256)
void cvt_f32_bf16_kernel(const float* __restrict__ x, __bf16* __restrict__ y, int n4) {
    int i = blockIdx.x * blockDim.x + threadIdx.x;
    if (i < n4) {
        float4 f = ((const float4*)x)[i];
        y[i * 4 + 0] = (__bf16)f.x;
        y[i * 4 + 1] = (__bf16)f.y;
        y[i * 4 + 2] = (__bf16)f.z;
        y[i * 4 + 3] = (__bf16)f.w;
    }
}

// ---------------------------------------------------------------------------
// Tiled GEMM: C[M,N] = A[M,K](bf16,row) x B[K,N](bf16,row) + bias
// WG = 256 threads = 8 waves (2x4), tile 128x128, K-step 32.
// Double-buffered LDS: async-stage tile t+1 while WMMAs consume tile t.
// ---------------------------------------------------------------------------
template <bool RELU, bool OUT_BF16>
__global__ __launch_bounds__(256)
void gemm_bf16_kernel(const __bf16* __restrict__ A, const __bf16* __restrict__ Bm,
                      const float* __restrict__ bias, void* __restrict__ Cout,
                      int M, int N, int K) {
    __shared__ __align__(16) __bf16 As[2][128][32];
    __shared__ __align__(16) __bf16 Bs[2][32][128];

    const int tid  = threadIdx.x;
    const int lane = tid & 31;
    const int wid  = tid >> 5;
    const int wm   = wid >> 2;        // 0..1  (64-row slab)
    const int wn   = wid & 3;         // 0..3  (32-col slab)
    const int l15  = lane & 15, half = lane >> 4;
    const int rowBase = blockIdx.y * 128;
    const int colBase = blockIdx.x * 128;

    auto stage = [&](int buf, int k0) {
#pragma unroll
        for (int i = 0; i < 2; ++i) {
            int q  = tid + i * 256;
            int ra = q >> 2, ca = (q & 3) * 8;
            copy16_g2l(&As[buf][ra][ca],
                       &A[(size_t)(rowBase + ra) * K + k0 + ca]);
            int rb = q >> 4, cb = (q & 15) * 8;
            copy16_g2l(&Bs[buf][rb][cb],
                       &Bm[(size_t)(k0 + rb) * N + colBase + cb]);
        }
    };

    f32x8 acc[4][2];
#pragma unroll
    for (int i = 0; i < 4; ++i)
#pragma unroll
        for (int j = 0; j < 2; ++j)
            acc[i][j] = (f32x8)F32X8_ZERO;

    stage(0, 0);
    async_wait0();
    __syncthreads();

    int buf = 0;
    for (int k0 = 0; k0 < K; k0 += 32) {
        if (k0 + 32 < K) stage(buf ^ 1, k0 + 32);     // issue-ahead (ASYNCcnt)
        if (k0 + 64 < K) {                            // global_prefetch_b8, 2 ahead
            int ra = tid >> 2, ca = (tid & 3) * 8;
            __builtin_prefetch(&A[(size_t)(rowBase + ra) * K + k0 + 64 + ca], 0, 1);
            int rb = tid >> 4, cb = (tid & 15) * 8;
            __builtin_prefetch(&Bm[(size_t)(k0 + 64 + rb) * N + colBase + cb], 0, 1);
        }

        bf16x16 af[4], bfr[2];
#pragma unroll
        for (int fm = 0; fm < 4; ++fm) {
            int r = wm * 64 + fm * 16 + l15;
#pragma unroll
            for (int e = 0; e < 16; ++e)
                af[fm][e] = As[buf][r][a_frag_k(half, e)];
        }
#pragma unroll
        for (int fn = 0; fn < 2; ++fn) {
            int c = wn * 32 + fn * 16 + l15;
#pragma unroll
            for (int e = 0; e < 16; ++e)
                bfr[fn][e] = Bs[buf][half * 16 + e][c];  // B[kk][n], kk = half*16+e
        }
#pragma unroll
        for (int fm = 0; fm < 4; ++fm)
#pragma unroll
            for (int fn = 0; fn < 2; ++fn)
                acc[fm][fn] = WMMA_BF16(af[fm], bfr[fn], acc[fm][fn]);

        async_wait0();        // next tile landed
        __syncthreads();      // all waves done reading buf / writing buf^1
        buf ^= 1;
    }

    // Epilogue: C layout lane l -> col l15, VGPR v -> row v + 8*half.
#pragma unroll
    for (int fm = 0; fm < 4; ++fm)
#pragma unroll
        for (int fn = 0; fn < 2; ++fn)
#pragma unroll
            for (int v = 0; v < 8; ++v) {
                int gr = rowBase + wm * 64 + fm * 16 + v + half * 8;
                int gc = colBase + wn * 32 + fn * 16 + l15;
                float val = acc[fm][fn][v];
                if (bias) val += bias[gc];
                if (RELU) val = fmaxf(val, 0.f);
                if (OUT_BF16) ((__bf16*)Cout)[(size_t)gr * N + gc] = (__bf16)val;
                else          ((float*)Cout)[(size_t)gr * N + gc] = val;
            }
}

// ---------------------------------------------------------------------------
// Flash attention. One wave per (16-query tile, head, batch).
// QKV: [6144][3072] bf16 (q | k | v per token row). ctx: [6144][1024] bf16.
// Gen cross-attention bias == "all pcpt keys + own gen key": flash loop over
// pcpt keys, then one analytic diagonal-key softmax update.
// K/V tiles double-buffered and staged with async-to-LDS copies.
// ---------------------------------------------------------------------------
__global__ __launch_bounds__(32)
void attn_kernel(const __bf16* __restrict__ QKV, __bf16* __restrict__ ctx,
                 int qBase0, int qStride, int kBase0, int kStride,
                 int nKeys, int isGen) {
    __shared__ __align__(16) __bf16 Qs[16][64];
    __shared__ __align__(16) __bf16 Ks[2][32][64];
    __shared__ __align__(16) __bf16 Vs[2][32][64];
    __shared__ __align__(16) __bf16 Ps[16][32];
    __shared__ float srow[16];

    const int lane = threadIdx.x & 31;
    const int l15  = lane & 15, half = lane >> 4;
    const int h    = blockIdx.y;
    const int b    = blockIdx.z;
    const int qBase = qBase0 + b * qStride + blockIdx.x * 16;
    const int kBase = kBase0 + b * kStride;
    const int ldq   = 3072;
    const float scale = 0.125f;   // HD^-0.5, HD=64

    auto stageKV = [&](int buf, int ktok) {     // ktok = global token row of key 0
#pragma unroll
        for (int i = 0; i < 8; ++i) {
            int q = lane + i * 32;
            int r = q >> 3, c = (q & 7) * 8;
            size_t rowOff = (size_t)(ktok + r) * ldq + h * 64 + c;
            copy16_g2l(&Ks[buf][r][c], &QKV[rowOff + 1024]);
            copy16_g2l(&Vs[buf][r][c], &QKV[rowOff + 2048]);
        }
    };

    // Stage Q tile (16x64) + first K/V tile
#pragma unroll
    for (int i = 0; i < 4; ++i) {
        int q = lane + i * 32;
        int r = q >> 3, c = (q & 7) * 8;
        copy16_g2l(&Qs[r][c], &QKV[(size_t)(qBase + r) * ldq + h * 64 + c]);
    }
    stageKV(0, kBase);
    async_wait0();
    __syncthreads();

    // Persistent A-fragments of Q (two K=32 slabs over HD=64)
    bf16x16 aq[2];
#pragma unroll
    for (int s = 0; s < 2; ++s)
#pragma unroll
        for (int e = 0; e < 16; ++e)
            aq[s][e] = Qs[l15][s * 32 + a_frag_k(half, e)];

    float m_i[8], l_i[8];
    f32x8 of[4];
#pragma unroll
    for (int v = 0; v < 8; ++v) { m_i[v] = -3.0e38f; l_i[v] = 0.f; }
#pragma unroll
    for (int f = 0; f < 4; ++f) of[f] = (f32x8)F32X8_ZERO;

    int buf = 0;
    for (int kt = 0; kt < nKeys; kt += 32) {
        if (kt + 32 < nKeys) stageKV(buf ^ 1, kBase + kt + 32);  // issue-ahead

        // Scores: S(16q x 32keys) = Q(16x64) . K^T; 2 frags x 2 K-slabs
        f32x8 s0 = (f32x8)F32X8_ZERO, s1 = (f32x8)F32X8_ZERO;
#pragma unroll
        for (int d = 0; d < 2; ++d) {
            bf16x16 bk0, bk1;
#pragma unroll
            for (int e = 0; e < 16; ++e) {
                int kk = d * 32 + half * 16 + e;        // reduction index (dim)
                bk0[e] = Ks[buf][l15][kk];              // keys 0..15
                bk1[e] = Ks[buf][16 + l15][kk];         // keys 16..31
            }
            s0 = WMMA_BF16(aq[d], bk0, s0);
            s1 = WMMA_BF16(aq[d], bk1, s1);
        }
        s0 *= scale;
        s1 *= scale;

        // Online softmax: row stats via 16-lane butterflies (rows live in halves)
        float mnew[8], psc[8];
#pragma unroll
        for (int v = 0; v < 8; ++v) {
            float x = fmaxf(s0[v], s1[v]);
#pragma unroll
            for (int off = 8; off >= 1; off >>= 1)
                x = fmaxf(x, __shfl_xor(x, off, 32));
            mnew[v] = fmaxf(m_i[v], x);
            psc[v]  = __expf(m_i[v] - mnew[v]);
        }
#pragma unroll
        for (int v = 0; v < 8; ++v) {
            float p0 = __expf(s0[v] - mnew[v]);
            float p1 = __expf(s1[v] - mnew[v]);
            s0[v] = p0; s1[v] = p1;
            float x = p0 + p1;
#pragma unroll
            for (int off = 8; off >= 1; off >>= 1)
                x += __shfl_xor(x, off, 32);
            l_i[v] = l_i[v] * psc[v] + x;
            m_i[v] = mnew[v];
        }
#pragma unroll
        for (int f = 0; f < 4; ++f)
#pragma unroll
            for (int v = 0; v < 8; ++v)
                of[f][v] *= psc[v];

        // Re-layout P (C-frag layout) -> A-frag layout via LDS
#pragma unroll
        for (int v = 0; v < 8; ++v) {
            Ps[v + 8 * half][l15]      = (__bf16)s0[v];
            Ps[v + 8 * half][16 + l15] = (__bf16)s1[v];
        }
        __syncthreads();
        bf16x16 pa;
#pragma unroll
        for (int e = 0; e < 16; ++e)
            pa[e] = Ps[l15][a_frag_k(half, e)];

        // O(16x64) += P(16x32) . V(32x64): 4 wmma over N
#pragma unroll
        for (int f = 0; f < 4; ++f) {
            bf16x16 bv;
#pragma unroll
            for (int e = 0; e < 16; ++e)
                bv[e] = Vs[buf][half * 16 + e][f * 16 + l15];
            of[f] = WMMA_BF16(pa, bv, of[f]);
        }

        async_wait0();          // next K/V tile landed
        __syncthreads();
        buf ^= 1;
    }

    if (isGen) {
        // Diagonal gen key: each query row attends its own token's k/v.
        stageKV(0, qBase);      // only rows 0..15 are used below
        async_wait0();
        __syncthreads();
        if (lane < 16) {
            float acc = 0.f;
            for (int d = 0; d < 64; ++d)
                acc += (float)Qs[lane][d] * (float)Ks[0][lane][d];
            srow[lane] = acc * scale;
        }
        __syncthreads();
#pragma unroll
        for (int v = 0; v < 8; ++v) {
            int r = v + 8 * half;
            float s  = srow[r];
            float mn = fmaxf(m_i[v], s);
            float sc = __expf(m_i[v] - mn);
            float pe = __expf(s - mn);
            l_i[v] = l_i[v] * sc + pe;
            m_i[v] = mn;
#pragma unroll
            for (int f = 0; f < 4; ++f)
                of[f][v] = of[f][v] * sc + pe * (float)Vs[0][r][f * 16 + l15];
        }
    }

    // Normalize + store ctx (bf16) for the out-projection GEMM
#pragma unroll
    for (int f = 0; f < 4; ++f)
#pragma unroll
        for (int v = 0; v < 8; ++v) {
            int r = v + 8 * half;
            float val = of[f][v] / l_i[v];
            ctx[(size_t)(qBase + r) * 1024 + h * 64 + f * 16 + l15] = (__bf16)val;
        }
}

// ---------------------------------------------------------------------------
// y = LayerNorm(x + residual) * gamma + beta ; writes f32 (+ optional bf16)
// One 256-thread WG per 1024-wide row.
// ---------------------------------------------------------------------------
__global__ __launch_bounds__(256)
void add_ln_kernel(const float* __restrict__ x,
                   const float* __restrict__ resLo, const float* __restrict__ resHi,
                   const float* __restrict__ gamma, const float* __restrict__ beta,
                   float* __restrict__ yf, __bf16* __restrict__ ybf) {
    __shared__ float red[256];
    const int row = blockIdx.x;
    const int tid = threadIdx.x;
    const float* res = (row < 4096) ? (resLo + (size_t)row * 1024)
                                    : (resHi + (size_t)(row - 4096) * 1024);
    const float* xr = x + (size_t)row * 1024;

    float v[4];
    float psum = 0.f;
#pragma unroll
    for (int k = 0; k < 4; ++k) {
        int j = tid + k * 256;
        v[k] = xr[j] + res[j];
        psum += v[k];
    }
    red[tid] = psum;
    __syncthreads();
    for (int off = 128; off > 0; off >>= 1) {
        if (tid < off) red[tid] += red[tid + off];
        __syncthreads();
    }
    float mean = red[0] * (1.f / 1024.f);
    __syncthreads();

    float pvar = 0.f;
#pragma unroll
    for (int k = 0; k < 4; ++k) { float d = v[k] - mean; pvar += d * d; }
    red[tid] = pvar;
    __syncthreads();
    for (int off = 128; off > 0; off >>= 1) {
        if (tid < off) red[tid] += red[tid + off];
        __syncthreads();
    }
    float rstd = rsqrtf(red[0] * (1.f / 1024.f) + 1e-5f);

#pragma unroll
    for (int k = 0; k < 4; ++k) {
        int j = tid + k * 256;
        float y = (v[k] - mean) * rstd * gamma[j] + beta[j];
        yf[(size_t)row * 1024 + j] = y;
        if (ybf) ybf[(size_t)row * 1024 + j] = (__bf16)y;
    }
}

// ---------------------------------------------------------------------------
// Orchestration
// ---------------------------------------------------------------------------
extern "C" void kernel_launch(void* const* d_in, const int* in_sizes, int n_in,
                              void* d_out, int out_size, void* d_ws, size_t ws_size,
                              hipStream_t stream) {
    (void)in_sizes; (void)n_in; (void)out_size; (void)ws_size;

    const float* pcpt   = (const float*)d_in[0];
    const float* gen    = (const float*)d_in[1];
    const float* Wqkv_w = (const float*)d_in[2];
    const float* Wqkv_b = (const float*)d_in[3];
    const float* out_w  = (const float*)d_in[4];
    const float* out_b  = (const float*)d_in[5];
    const float* l1_w   = (const float*)d_in[6];
    const float* l1_b   = (const float*)d_in[7];
    const float* l2_w   = (const float*)d_in[8];
    const float* l2_b   = (const float*)d_in[9];
    const float* n1_g   = (const float*)d_in[10];
    const float* n1_b   = (const float*)d_in[11];
    const float* n2_g   = (const float*)d_in[12];
    const float* n2_b   = (const float*)d_in[13];

    char* ws = (char*)d_ws;
    size_t off = 0;
    auto carve = [&](size_t bytes) -> char* {
        char* p = ws + off;
        off += (bytes + 255) & ~(size_t)255;
        return p;
    };

    // Rows 0..4095 = pcpt (b-major), rows 4096..6143 = gen : matches d_out order.
    __bf16* wqkv_bf = (__bf16*)carve((size_t)1024 * 3072 * 2);
    __bf16* wo_bf   = (__bf16*)carve((size_t)1024 * 1024 * 2);
    __bf16* w1_bf   = (__bf16*)carve((size_t)1024 * 4096 * 2);
    __bf16* w2_bf   = (__bf16*)carve((size_t)4096 * 1024 * 2);
    __bf16* xcat_bf = (__bf16*)carve((size_t)6144 * 1024 * 2);
    __bf16* qkv_bf  = (__bf16*)carve((size_t)6144 * 3072 * 2);
    __bf16* ctx_bf  = (__bf16*)carve((size_t)6144 * 1024 * 2);
    float*  proj_f  = (float*) carve((size_t)6144 * 1024 * 4);
    float*  h1_f    = (float*) carve((size_t)6144 * 1024 * 4);
    __bf16* h1_bf   = (__bf16*)carve((size_t)6144 * 1024 * 2);
    __bf16* mid_bf  = (__bf16*)carve((size_t)6144 * 4096 * 2);
    float*  ffn2_f  = (float*) carve((size_t)6144 * 1024 * 4);

    auto cvt = [&](const float* src, __bf16* dst, int n) {
        int n4 = n / 4;
        cvt_f32_bf16_kernel<<<(n4 + 255) / 256, 256, 0, stream>>>(src, dst, n4);
    };

    // 1) bf16 staging
    cvt(Wqkv_w, wqkv_bf, 1024 * 3072);
    cvt(out_w,  wo_bf,   1024 * 1024);
    cvt(l1_w,   w1_bf,   1024 * 4096);
    cvt(l2_w,   w2_bf,   4096 * 1024);
    cvt(pcpt,   xcat_bf,                       2 * 2048 * 1024);
    cvt(gen,    xcat_bf + (size_t)4096 * 1024, 2 * 1024 * 1024);

    // 2) QKV projection: [6144,1024] x [1024,3072] -> bf16
    gemm_bf16_kernel<false, true><<<dim3(3072 / 128, 6144 / 128), 256, 0, stream>>>(
        xcat_bf, wqkv_bf, Wqkv_b, qkv_bf, 6144, 3072, 1024);

    // 3) Attention. pcpt self-attn: Q,K,V in pcpt rows of same batch.
    attn_kernel<<<dim3(2048 / 16, 16, 2), 32, 0, stream>>>(
        qkv_bf, ctx_bf, /*q*/0, 2048, /*k*/0, 2048, /*nKeys*/2048, /*isGen*/0);
    //    gen cross-attn: keys = pcpt rows of same batch + own diagonal gen key.
    attn_kernel<<<dim3(1024 / 16, 16, 2), 32, 0, stream>>>(
        qkv_bf, ctx_bf, /*q*/4096, 1024, /*k*/0, 2048, /*nKeys*/2048, /*isGen*/1);

    // 4) Out projection -> f32
    gemm_bf16_kernel<false, false><<<dim3(1024 / 128, 6144 / 128), 256, 0, stream>>>(
        ctx_bf, wo_bf, out_b, proj_f, 6144, 1024, 1024);

    // 5) LN1(x + ctx_proj) -> h1 (f32 + bf16)
    add_ln_kernel<<<6144, 256, 0, stream>>>(proj_f, pcpt, gen, n1_g, n1_b, h1_f, h1_bf);

    // 6) FFN1 (+ReLU, bf16 out)
    gemm_bf16_kernel<true, true><<<dim3(4096 / 128, 6144 / 128), 256, 0, stream>>>(
        h1_bf, w1_bf, l1_b, mid_bf, 6144, 4096, 1024);

    // 7) FFN2 -> f32
    gemm_bf16_kernel<false, false><<<dim3(1024 / 128, 6144 / 128), 256, 0, stream>>>(
        mid_bf, w2_bf, l2_b, ffn2_f, 6144, 1024, 4096);

    // 8) LN2(h1 + ffn) -> d_out (p then g, already in row order)
    add_ln_kernel<<<6144, 256, 0, stream>>>(
        ffn2_f, h1_f, h1_f + (size_t)4096 * 1024, n2_g, n2_b, (float*)d_out, nullptr);
}
